// EncoderLayer_377957122174
// MI455X (gfx1250) — matrix operand order repaired and checked by reading, not compiled, not run on hardware
//
#include <hip/hip_runtime.h>

// ---------------------------------------------------------------------------
// CDNA5 (gfx1250, wave32) transformer encoder layer.
// - All GEMM-class math through v_wmma_f32_16x16x32_bf16.
// - HBM -> LDS staging via global_load_async_to_lds_b128 (ASYNCcnt),
//   double-buffered so staging overlaps the matrix pipeline.
// ---------------------------------------------------------------------------

typedef __bf16 v16bf __attribute__((ext_vector_type(16)));
typedef float  v8f   __attribute__((ext_vector_type(8)));

union Frag16 {
  v16bf v;
  unsigned int   u[8];
  unsigned short s[16];
};

__device__ __forceinline__ unsigned short f2bf(float f) {
  __bf16 h = (__bf16)f;
  return __builtin_bit_cast(unsigned short, h);
}

__device__ __forceinline__ v8f wmma_bf16(v16bf a, v16bf b, v8f c) {
  return __builtin_amdgcn_wmma_f32_16x16x32_bf16(
      /*neg_a=*/false, a, /*neg_b=*/false, b,
      /*c_mod=*/(short)0, c, /*reuse_a=*/false, /*reuse_b=*/false);
}

// LDS byte offset of a __shared__ object: flat LDS aperture keeps the
// LDS offset in addr[31:0] (CDNA5 ISA 10.2 aperture mapping).
__device__ __forceinline__ unsigned lds_off(const void* p) {
  return (unsigned)(unsigned long long)p;
}

// Async HBM->LDS copy of 16 bytes per lane; tracked with ASYNCcnt.
__device__ __forceinline__ void async_ld_b128(unsigned lds_byte_off,
                                              const void* gaddr) {
  asm volatile("global_load_async_to_lds_b128 %0, %1, off"
               :: "v"(lds_byte_off), "v"(gaddr)
               : "memory");
}

__device__ __forceinline__ void wait_async0() {
  asm volatile("s_wait_asynccnt 0" ::: "memory");
}

// A-matrix 16x32 bf16 fragment from LDS tile stored [row][k] (row-major,
// stride in u16 elements). Contiguous-k layout (not ISA table order): lane m=l&15, half=l>>4;
// VGPR v holds K = kbase + 16*half + 2v, pairs contiguous.
__device__ __forceinline__ v16bf load_a_frag(const unsigned short* lds,
                                             int row, int stride,
                                             int kbase, int hf) {
  Frag16 f;
  const int base = row * stride + kbase + 16 * hf;
#pragma unroll
  for (int v = 0; v < 8; ++v) {
    const int k = 2 * v;
    f.u[v] = *reinterpret_cast<const unsigned int*>(lds + base + k);
  }
  return f.v;
}

// B-matrix 32x16 bf16 fragment; source LDS tile stored [k][n] (stride in u16).
// Layout: lane n=l&15, half=l>>4; element e maps to K = kbase + 16*half + e.
__device__ __forceinline__ v16bf load_b_frag_kn(const unsigned short* lds,
                                                int col, int stride,
                                                int kbase, int hf) {
  Frag16 f;
  const int k0 = kbase + 16 * hf;
#pragma unroll
  for (int e = 0; e < 16; ++e) f.s[e] = lds[(k0 + e) * stride + col];
  return f.v;
}

// B-matrix fragment where the LDS tile is stored [n][k] (K contiguous):
// dword reads of K-pairs.
__device__ __forceinline__ v16bf load_b_frag_nk(const unsigned short* lds,
                                                int nrow, int stride,
                                                int kbase, int hf) {
  Frag16 f;
  const int base = nrow * stride + kbase + 16 * hf;
#pragma unroll
  for (int v = 0; v < 8; ++v)
    f.u[v] = *reinterpret_cast<const unsigned int*>(lds + base + 2 * v);
  return f.v;
}

// ---------------------------------------------------------------------------
// fp32 -> bf16 conversion (weights)
// ---------------------------------------------------------------------------
__global__ void cvt_bf16_kernel(const float* __restrict__ in,
                                unsigned short* __restrict__ out, int n) {
  int i = blockIdx.x * blockDim.x + threadIdx.x;
  const int stride = gridDim.x * blockDim.x;
  for (; i < n; i += stride) out[i] = f2bf(in[i]);
}

// ---------------------------------------------------------------------------
// LayerNorm (torch flavor: alpha*(x-mu)/(std_ddof1 + eps) + beta), bf16 out.
// One 256-thread block per row of 1024.
// ---------------------------------------------------------------------------
__global__ __launch_bounds__(256)
void layernorm_kernel(const float* __restrict__ x,
                      const float* __restrict__ alpha,
                      const float* __restrict__ beta,
                      unsigned short* __restrict__ out) {
  __shared__ float red[256];
  __shared__ float smu, ssd;
  const int row = blockIdx.x, t = threadIdx.x;
  const float* xr = x + (size_t)row * 1024;
  float v[4];
  float s = 0.f, ss = 0.f;
#pragma unroll
  for (int i = 0; i < 4; ++i) {
    v[i] = xr[t + i * 256];
    s += v[i];
    ss += v[i] * v[i];
  }
  red[t] = s;
  __syncthreads();
#pragma unroll
  for (int off = 128; off > 0; off >>= 1) {
    if (t < off) red[t] += red[t + off];
    __syncthreads();
  }
  if (t == 0) smu = red[0] * (1.0f / 1024.0f);
  __syncthreads();
  red[t] = ss;
  __syncthreads();
#pragma unroll
  for (int off = 128; off > 0; off >>= 1) {
    if (t < off) red[t] += red[t + off];
    __syncthreads();
  }
  if (t == 0) {
    const float mu = smu;
    const float var = (red[0] - 1024.0f * mu * mu) * (1.0f / 1023.0f);
    ssd = sqrtf(fmaxf(var, 0.0f));
  }
  __syncthreads();
  const float mu = smu;
  const float inv = 1.0f / (ssd + 1e-6f);
#pragma unroll
  for (int i = 0; i < 4; ++i) {
    const int c = t + i * 256;
    out[(size_t)row * 1024 + c] = f2bf(alpha[c] * (v[i] - mu) * inv + beta[c]);
  }
}

// ---------------------------------------------------------------------------
// bf16 WMMA GEMM: out[M,N] = A[M,K] * B[K,N] (+bias) (relu) (+resid).
// 256 threads = 8 waves (2x4), 128x128 tile, BK=64 slabs.
// Double-buffered LDS with async HBM->LDS staging (ASYNCcnt).
// Each wave: 64x32 sub-tile = 4x2 accumulators, 16 WMMA per K-slab.
// ---------------------------------------------------------------------------
template <bool BIAS, bool RELU, bool RESID, bool OUTBF>
__global__ __launch_bounds__(256)
void gemm_bf16_kernel(const unsigned short* __restrict__ A,
                      const unsigned short* __restrict__ Bm,
                      const float* __restrict__ bias,
                      const float* __restrict__ resid,
                      void* __restrict__ outp, int M, int N, int Kd) {
  __shared__ __align__(16) unsigned short sA[2][128 * 72];   // [m][k], pad 72
  __shared__ __align__(16) unsigned short sB[2][64 * 136];   // [k][n], pad 136
  const int tid = threadIdx.x;
  const int wid = tid >> 5, lane = tid & 31;
  const int l15 = lane & 15, hf = lane >> 4;
  const int wm = wid >> 2, wn = wid & 3;
  const int bm = blockIdx.y * 128, bn = blockIdx.x * 128;

  v8f acc[4][2];
#pragma unroll
  for (int mt = 0; mt < 4; ++mt)
#pragma unroll
    for (int nt = 0; nt < 2; ++nt)
#pragma unroll
      for (int r = 0; r < 8; ++r) acc[mt][nt][r] = 0.0f;

  const int ra = tid >> 3, ca = (tid & 7) << 3;
  const int rb = tid >> 4, cb = (tid & 15) << 3;

  auto issue_slab = [&](int k0, int buf) {
#pragma unroll
    for (int p = 0; p < 4; ++p) {  // A: 128x64
      const int row = p * 32 + ra;
      async_ld_b128(lds_off(&sA[buf][row * 72 + ca]),
                    A + (size_t)(bm + row) * Kd + k0 + ca);
    }
#pragma unroll
    for (int p = 0; p < 4; ++p) {  // B: 64x128
      const int kr = p * 16 + rb;
      async_ld_b128(lds_off(&sB[buf][kr * 136 + cb]),
                    Bm + (size_t)(k0 + kr) * N + bn + cb);
    }
  };

  const int nslab = Kd >> 6;
  issue_slab(0, 0);

  for (int i = 0; i < nslab; ++i) {
    wait_async0();      // my async writes for slab i have landed in LDS
    __syncthreads();    // everyone's have; everyone done reading buf (i+1)&1
    if (i + 1 < nslab) issue_slab((i + 1) << 6, (i + 1) & 1);

    const unsigned short* cA = sA[i & 1];
    const unsigned short* cB = sB[i & 1];
#pragma unroll
    for (int ks = 0; ks < 2; ++ks) {
      v16bf af[4], bf[2];
#pragma unroll
      for (int mt = 0; mt < 4; ++mt)
        af[mt] = load_a_frag(cA, wm * 64 + mt * 16 + l15, 72, ks * 32, hf);
#pragma unroll
      for (int nt = 0; nt < 2; ++nt)
        bf[nt] = load_b_frag_kn(cB, wn * 32 + nt * 16 + l15, 136, ks * 32, hf);
#pragma unroll
      for (int mt = 0; mt < 4; ++mt)
#pragma unroll
        for (int nt = 0; nt < 2; ++nt)
          acc[mt][nt] = wmma_bf16(af[mt], bf[nt], acc[mt][nt]);
    }
  }

#pragma unroll
  for (int nt = 0; nt < 2; ++nt) {
    const int col = bn + wn * 32 + nt * 16 + l15;
    const float bv = BIAS ? bias[col] : 0.0f;
#pragma unroll
    for (int mt = 0; mt < 4; ++mt) {
#pragma unroll
      for (int r = 0; r < 8; ++r) {
        const int row = bm + wm * 64 + mt * 16 + r + 8 * hf;
        float v = acc[mt][nt][r] + bv;
        if (RELU) v = fmaxf(v, 0.0f);
        if (RESID) v += resid[(size_t)row * N + col];
        if (OUTBF)
          ((unsigned short*)outp)[(size_t)row * N + col] = f2bf(v);
        else
          ((float*)outp)[(size_t)row * N + col] = v;
      }
    }
  }
}

// ---------------------------------------------------------------------------
// Flash attention: one block = (b, h, 128-query tile). 8 waves x 16 rows.
// K/V chunks streamed with async HBM->LDS double buffering; online softmax
// with per-row running max/sum; P staged via per-wave LDS to convert the
// score D-layout into the A-fragment layout for P*V.
// ---------------------------------------------------------------------------
__global__ __launch_bounds__(256)
void attn_kernel(const unsigned short* __restrict__ Q,
                 const unsigned short* __restrict__ Kx,
                 const unsigned short* __restrict__ V,
                 const int* __restrict__ mask,
                 unsigned short* __restrict__ Cc) {
  __shared__ __align__(16) unsigned short sQ[128 * 72];        // [q][dk]
  __shared__ __align__(16) unsigned short sK[2][64 * 72];      // [key][dk]
  __shared__ __align__(16) unsigned short sV[2][64 * 72];      // [key][dk]
  __shared__ __align__(16) unsigned short sP[8 * 16 * 72];     // per-wave [m][key]

  const int tid = threadIdx.x;
  const int wid = tid >> 5, lane = tid & 31;
  const int l15 = lane & 15, hf = lane >> 4;
  const int bidx = blockIdx.x;
  const int b = bidx >> 7, h = (bidx >> 3) & 15, qt = bidx & 7;
  const size_t rowbase = (size_t)b * 1024 + qt * 128;
  const int colbase = h * 64;

  const int r8 = tid >> 3, c8 = (tid & 7) << 3;

  auto issue_kv = [&](int kc, int buf) {
#pragma unroll
    for (int p = 0; p < 2; ++p) {  // 64 keys x 64 dk, K and V
      const int row = p * 32 + r8;
      const size_t g = ((size_t)b * 1024 + kc * 64 + row) * 1024 + colbase + c8;
      async_ld_b128(lds_off(&sK[buf][row * 72 + c8]), Kx + g);
      async_ld_b128(lds_off(&sV[buf][row * 72 + c8]), V + g);
    }
  };

  {  // stage Q tile 128x64 (async) + first K/V chunk
#pragma unroll
    for (int p = 0; p < 4; ++p) {
      const int row = p * 32 + r8;
      async_ld_b128(lds_off(&sQ[row * 72 + c8]),
                    Q + (rowbase + row) * 1024 + colbase + c8);
    }
    issue_kv(0, 0);
  }

  v16bf qa[2];
  v8f o[4];
  float rm[8], rl[8];
#pragma unroll
  for (int nt = 0; nt < 4; ++nt)
#pragma unroll
    for (int r = 0; r < 8; ++r) o[nt][r] = 0.0f;
#pragma unroll
  for (int r = 0; r < 8; ++r) { rm[r] = -1e30f; rl[r] = 0.0f; }

  const int* mrow = mask + b * 1024;

  for (int kc = 0; kc < 16; ++kc) {
    wait_async0();
    __syncthreads();
    if (kc == 0) {
#pragma unroll
      for (int ks = 0; ks < 2; ++ks)
        qa[ks] = load_a_frag(sQ, wid * 16 + l15, 72, ks * 32, hf);
    }
    if (kc + 1 < 16) issue_kv(kc + 1, (kc + 1) & 1);

    const unsigned short* cK = sK[kc & 1];
    const unsigned short* cV = sV[kc & 1];

    // scores: S[16q x 64key] = Q * K^T  (B matrix = K stored [key][dk] -> nk)
    v8f sacc[4];
#pragma unroll
    for (int nt = 0; nt < 4; ++nt) {
#pragma unroll
      for (int r = 0; r < 8; ++r) sacc[nt][r] = 0.0f;
#pragma unroll
      for (int ks = 0; ks < 2; ++ks) {
        const v16bf kb = load_b_frag_nk(cK, nt * 16 + l15, 72, ks * 32, hf);
        sacc[nt] = wmma_bf16(qa[ks], kb, sacc[nt]);
      }
    }

    // scale + mask (each lane owns a fixed key column per n-tile)
    float sf[4][8];
#pragma unroll
    for (int nt = 0; nt < 4; ++nt) {
      const int key = kc * 64 + nt * 16 + l15;
      const bool mk = mrow[key] != 0;
#pragma unroll
      for (int r = 0; r < 8; ++r)
        sf[nt][r] = mk ? sacc[nt][r] * 0.125f : -1e9f;
    }

    // per-row chunk max (reduce across the 16 lanes of each half-wave)
    float cm[8];
#pragma unroll
    for (int r = 0; r < 8; ++r)
      cm[r] = fmaxf(fmaxf(sf[0][r], sf[1][r]), fmaxf(sf[2][r], sf[3][r]));
#pragma unroll
    for (int off = 1; off < 16; off <<= 1)
#pragma unroll
      for (int r = 0; r < 8; ++r)
        cm[r] = fmaxf(cm[r], __shfl_xor(cm[r], off, 32));

    float al[8];
#pragma unroll
    for (int r = 0; r < 8; ++r) {
      const float nm = fmaxf(rm[r], cm[r]);
      al[r] = __expf(rm[r] - nm);
      rm[r] = nm;
    }

    float cl[8];
#pragma unroll
    for (int r = 0; r < 8; ++r) cl[r] = 0.0f;
#pragma unroll
    for (int nt = 0; nt < 4; ++nt)
#pragma unroll
      for (int r = 0; r < 8; ++r) {
        const float p = __expf(sf[nt][r] - rm[r]);
        sf[nt][r] = p;
        cl[r] += p;
      }
#pragma unroll
    for (int off = 1; off < 16; off <<= 1)
#pragma unroll
      for (int r = 0; r < 8; ++r) cl[r] += __shfl_xor(cl[r], off, 32);
#pragma unroll
    for (int r = 0; r < 8; ++r) rl[r] = rl[r] * al[r] + cl[r];
#pragma unroll
    for (int nt = 0; nt < 4; ++nt)
#pragma unroll
      for (int r = 0; r < 8; ++r) o[nt][r] *= al[r];

    // stage P (per-wave buffer): D-layout (col=key per lane) -> [m][key]
#pragma unroll
    for (int nt = 0; nt < 4; ++nt)
#pragma unroll
      for (int r = 0; r < 8; ++r)
        sP[(wid * 16 + r + 8 * hf) * 72 + nt * 16 + l15] = f2bf(sf[nt][r]);
    asm volatile("s_wait_dscnt 0" ::: "memory");  // wave-local LDS RAW

    v16bf pa[2];
#pragma unroll
    for (int ks = 0; ks < 2; ++ks)
      pa[ks] = load_a_frag(sP + wid * 16 * 72, l15, 72, ks * 32, hf);

    // O += P * V   (B matrix = V stored [key][dk] -> kn, col = dk)
#pragma unroll
    for (int nt = 0; nt < 4; ++nt)
#pragma unroll
      for (int ks = 0; ks < 2; ++ks) {
        const v16bf vb = load_b_frag_kn(cV, nt * 16 + l15, 72, ks * 32, hf);
        o[nt] = wmma_bf16(pa[ks], vb, o[nt]);
      }
  }

  // normalize and write concat [B,S,D] at column block h*64
#pragma unroll
  for (int nt = 0; nt < 4; ++nt) {
    const int col = colbase + nt * 16 + l15;
#pragma unroll
    for (int r = 0; r < 8; ++r) {
      const int row = wid * 16 + r + 8 * hf;
      Cc[(rowbase + row) * 1024 + col] = f2bf(o[nt][r] / rl[r]);
    }
  }
}

// ---------------------------------------------------------------------------
// Host-side orchestration
// ---------------------------------------------------------------------------
extern "C" void kernel_launch(void* const* d_in, const int* in_sizes, int n_in,
                              void* d_out, int out_size, void* d_ws,
                              size_t ws_size, hipStream_t stream) {
  const float* x   = (const float*)d_in[0];
  const int*  mask = (const int*)d_in[1];
  const float* n1a = (const float*)d_in[2];
  const float* n1b = (const float*)d_in[3];
  const float* n2a = (const float*)d_in[4];
  const float* n2b = (const float*)d_in[5];
  const float* wq = (const float*)d_in[6];   const float* bq = (const float*)d_in[7];
  const float* wk = (const float*)d_in[8];   const float* bk = (const float*)d_in[9];
  const float* wv = (const float*)d_in[10];  const float* bv = (const float*)d_in[11];
  const float* wo = (const float*)d_in[12];  const float* bo = (const float*)d_in[13];
  const float* w1 = (const float*)d_in[14];  const float* b1 = (const float*)d_in[15];
  const float* w2 = (const float*)d_in[16];  const float* b2 = (const float*)d_in[17];
  float* out = (float*)d_out;
  char* ws = (char*)d_ws;

  const size_t M = 8192;  // B*S
  // workspace layout (bytes)
  unsigned short* X2  = (unsigned short*)(ws + 0);            // 16 MiB bf16 [8192,1024]
  unsigned short* WQB = (unsigned short*)(ws + (16u << 20));
  unsigned short* WKB = (unsigned short*)(ws + (18u << 20));
  unsigned short* WVB = (unsigned short*)(ws + (20u << 20));
  unsigned short* WOB = (unsigned short*)(ws + (22u << 20));
  unsigned short* W1B = (unsigned short*)(ws + (24u << 20));  // 4 MiB
  unsigned short* W2B = (unsigned short*)(ws + (28u << 20));  // 4 MiB
  unsigned short* Qb  = (unsigned short*)(ws + (32u << 20));  // 16 MiB
  unsigned short* Kb  = (unsigned short*)(ws + (48u << 20));  // 16 MiB
  unsigned short* Vb  = (unsigned short*)(ws + (64u << 20));  // 16 MiB
  unsigned short* Cc  = (unsigned short*)(ws + (80u << 20));  // 16 MiB
  unsigned short* Hb  = (unsigned short*)(ws + (32u << 20));  // 32 MiB, aliases Q+K (dead)

  // 1) weights -> bf16
  cvt_bf16_kernel<<<2048, 256, 0, stream>>>(wq, WQB, 1024 * 1024);
  cvt_bf16_kernel<<<2048, 256, 0, stream>>>(wk, WKB, 1024 * 1024);
  cvt_bf16_kernel<<<2048, 256, 0, stream>>>(wv, WVB, 1024 * 1024);
  cvt_bf16_kernel<<<2048, 256, 0, stream>>>(wo, WOB, 1024 * 1024);
  cvt_bf16_kernel<<<2048, 256, 0, stream>>>(w1, W1B, 1024 * 2048);
  cvt_bf16_kernel<<<2048, 256, 0, stream>>>(w2, W2B, 2048 * 1024);

  // 2) LN1
  layernorm_kernel<<<(int)M, 256, 0, stream>>>(x, n1a, n1b, X2);

  // 3) Q/K/V projections
  dim3 g1024(1024 / 128, (int)M / 128);
  gemm_bf16_kernel<true, false, false, true><<<g1024, 256, 0, stream>>>(
      X2, WQB, bq, nullptr, Qb, (int)M, 1024, 1024);
  gemm_bf16_kernel<true, false, false, true><<<g1024, 256, 0, stream>>>(
      X2, WKB, bk, nullptr, Kb, (int)M, 1024, 1024);
  gemm_bf16_kernel<true, false, false, true><<<g1024, 256, 0, stream>>>(
      X2, WVB, bv, nullptr, Vb, (int)M, 1024, 1024);

  // 4) attention -> concat
  attn_kernel<<<8 * 16 * 8, 256, 0, stream>>>(Qb, Kb, Vb, mask, Cc);

  // 5) output projection + residual(x) -> d_out (fp32)
  gemm_bf16_kernel<true, false, true, false><<<g1024, 256, 0, stream>>>(
      Cc, WOB, bo, x, out, (int)M, 1024, 1024);

  // 6) LN2 on d_out -> X2 (bf16)
  layernorm_kernel<<<(int)M, 256, 0, stream>>>(out, n2a, n2b, X2);

  // 7) FFN1 (relu) -> Hb (bf16, 8192x2048)
  dim3 g2048(2048 / 128, (int)M / 128);
  gemm_bf16_kernel<true, true, false, true><<<g2048, 256, 0, stream>>>(
      X2, W1B, b1, nullptr, Hb, (int)M, 2048, 1024);

  // 8) FFN2 + residual(d_out) -> d_out (fp32)
  gemm_bf16_kernel<true, false, true, false><<<g1024, 256, 0, stream>>>(
      Hb, W2B, b2, out, out, (int)M, 1024, 2048);
}
